// TextMMadapter_45861660786839
// MI455X (gfx1250) — compile-verified, matrix operand order checked by reading, not run on hardware
//
#include <hip/hip_runtime.h>

// ---------------------------------------------------------------------------
// Sizes (compile-time constants from the reference)
// ---------------------------------------------------------------------------
#define SEQ      2048
#define BATCH    4
#define HID      4096
#define DM       128
#define DIN      256
#define DST      16
#define HD       64
#define NH       4
#define DCONV    7
#define CONV_DIM 288            // DIN + 2*DST
#define DPROJ    552            // 2*DIN + 2*DST + 2*NH
#define DPROJ_P  576            // padded to multiple of 64 for GEMM tiles
#define NTOK     (SEQ * BATCH)  // 8192 tokens

typedef __attribute__((ext_vector_type(16))) __bf16        v16bf;
typedef __attribute__((ext_vector_type(8)))  float         v8f;
typedef __attribute__((ext_vector_type(4)))  unsigned int  v4u;
typedef __attribute__((ext_vector_type(8)))  int           v8i;
typedef __attribute__((ext_vector_type(4)))  int           v4i;

#if defined(__has_builtin)
#if __has_builtin(__builtin_amdgcn_tensor_load_to_lds) && \
    __has_builtin(__builtin_amdgcn_s_wait_tensorcnt)
#define USE_TDM 1
#endif
#endif

union FragBF {
    v4u  q[2];
    v16bf v;
};

__device__ __forceinline__ unsigned short f2bf(float f) {
    unsigned int u = __float_as_uint(f);
    unsigned int r = u + 0x7FFFu + ((u >> 16) & 1u);   // round-to-nearest-even
    return (unsigned short)(r >> 16);
}

__device__ __forceinline__ float gelu_exact(float v) {
    return 0.5f * v * (1.0f + erff(v * 0.70710678118654752f));
}
__device__ __forceinline__ float silu(float v) {
    return v / (1.0f + expf(-v));
}
__device__ __forceinline__ float softplus(float v) {
    return (v > 20.0f) ? v : log1pf(expf(v));
}

#ifdef USE_TDM
// Issue one 2D TDM tile load: (tile1 rows x 32 cols) of 2-byte elements from a
// row-major tensor with row length `rowlen`, into LDS at `ldst`.
// D# layout per CDNA5 ISA ch.8 (group0: count/lds/global/type, group1: dims).
__device__ __forceinline__ void tdm_tile_load(const unsigned short* gsrc,
                                              unsigned short* ldst,
                                              unsigned int rowlen,
                                              unsigned int nrows,
                                              unsigned int tile1) {
    unsigned long long ga = (unsigned long long)(const void*)gsrc;
    unsigned int lds = (unsigned int)(unsigned long long)(void*)ldst;
    v4u g0;
    g0.x = 1u;                                        // count = 1
    g0.y = lds;                                       // lds_addr
    g0.z = (unsigned int)ga;                          // global_addr[31:0]
    g0.w = (unsigned int)(ga >> 32) | (2u << 30);     // addr[56:32] | type = 2
    v8i g1;
    g1[0] = (int)(1u << 16);                          // data_size = 1 (2 bytes)
    g1[1] = (int)((rowlen & 0xFFFFu) << 16);          // tensor_dim0 lo16
    g1[2] = (int)((rowlen >> 16) | ((nrows & 0xFFFFu) << 16)); // dim0 hi | dim1 lo
    g1[3] = (int)((nrows >> 16) | (32u << 16));       // dim1 hi | tile_dim0 = 32
    g1[4] = (int)tile1;                               // tile_dim1
    g1[5] = (int)rowlen;                              // tensor_dim0_stride
    g1[6] = 0;
    g1[7] = 0;
    v4i gz  = {0, 0, 0, 0};
    v8i gz8 = {0, 0, 0, 0, 0, 0, 0, 0};
    __builtin_amdgcn_tensor_load_to_lds(g0, g1, gz, gz, gz8, 0);
}
#endif

// ---------------------------------------------------------------------------
// Tiled bf16 WMMA GEMM:  C[M,N] = A[M,K] * W[N,K]^T  (+bias, optional epilogue)
//   block tile: 128 (M) x 64 (N), 256 threads = 8 waves, K step 32.
//   Both A and W tiles are DMA'd by the Tensor Data Mover into double-buffered
//   LDS; the TDM fill of tile i+1 overlaps the WMMA work on tile i.
//   epilogue 0: C[r*ldc + c] = acc + bias
//   epilogue 1: out[(l*BATCH+b)*HID + c] = resid[...] + acc + bias  (r = b*SEQ+l)
// ---------------------------------------------------------------------------
__global__ __launch_bounds__(256)
void gemm_bf16_kernel(const unsigned short* __restrict__ A,
                      const unsigned short* __restrict__ W,
                      const float* __restrict__ bias,
                      float* __restrict__ C,
                      int M, int K, int ldc, int Nrows, int epilogue,
                      const float* __restrict__ resid) {
    __shared__ __align__(16) unsigned short As[2][128 * 32];
    __shared__ __align__(16) unsigned short Ws[2][64 * 32];

    const int tid  = threadIdx.x;
    const int wave = tid >> 5;
    const int lane = tid & 31;
    const int ml   = lane & 15;
    const int hi   = lane >> 4;

    const int m0 = blockIdx.x * 128;
    const int n0 = blockIdx.y * 64;

    v8f acc[4] = {};

    const int nIter = K >> 5;   // K / 32

#ifdef USE_TDM
    if (tid == 0) {             // prologue: DMA first tile pair into buffer 0
        tdm_tile_load(A + (size_t)m0 * K, &As[0][0], (unsigned)K, (unsigned)M, 128u);
        tdm_tile_load(W + (size_t)n0 * K, &Ws[0][0], (unsigned)K, (unsigned)Nrows, 64u);
    }
#endif

    for (int it = 0; it < nIter; ++it) {
        const int buf = it & 1;
        const int kb  = it << 5;

#ifdef USE_TDM
        __syncthreads();   // all waves done reading buf^1 (iteration it-1)
        if (it + 1 < nIter) {
            if (tid == 0) {   // prefetch next tile pair into the other buffer
                tdm_tile_load(A + (size_t)m0 * K + kb + 32, &As[buf ^ 1][0],
                              (unsigned)K, (unsigned)M, 128u);
                tdm_tile_load(W + (size_t)n0 * K + kb + 32, &Ws[buf ^ 1][0],
                              (unsigned)K, (unsigned)Nrows, 64u);
            }
            // current pair complete when <= 2 (the just-issued pair) outstanding
            __builtin_amdgcn_s_wait_tensorcnt(2);
        } else {
            __builtin_amdgcn_s_wait_tensorcnt(0);
        }
        __syncthreads();   // publish TDM-filled buf to all waves
#else
        __syncthreads();
        {   // fallback: synchronous staging through VGPRs
            int row = tid >> 2, q = tid & 3;
            ((v4u*)&Ws[buf][0])[tid] =
                *(const v4u*)(W + (size_t)(n0 + row) * K + kb + q * 8);
            #pragma unroll
            for (int j = 0; j < 2; j++) {
                int chunk = tid + j * 256;
                int r2 = chunk >> 2, q2 = chunk & 3;
                ((v4u*)&As[buf][0])[chunk] =
                    *(const v4u*)(A + (size_t)(m0 + r2) * K + kb + q2 * 8);
            }
        }
        __syncthreads();
#endif

        // A fragment (16x32 bf16 layout: lanes 0-15 K=0..7,16..23;
        // lanes 16-31 K=8..15,24..31)
        FragBF af;
        const unsigned short* abase = &As[buf][0] + (wave * 16 + ml) * 32 + hi * 8;
        af.q[0] = *(const v4u*)(abase);
        af.q[1] = *(const v4u*)(abase + 16);

        #pragma unroll
        for (int nb = 0; nb < 4; nb++) {
            FragBF bf;
            const unsigned short* bbase = &Ws[buf][0] + (nb * 16 + ml) * 32 + hi * 8;
            bf.q[0] = *(const v4u*)(bbase);
            bf.q[1] = *(const v4u*)(bbase + 16);
            acc[nb] = __builtin_amdgcn_wmma_f32_16x16x32_bf16(
                false, af.v, false, bf.v, (short)0, acc[nb], false, false);
        }
    }

    // C tile: lanes 0-15 hold M = v (+wave base), lanes 16-31 M = v+8; N = ml
    #pragma unroll
    for (int nb = 0; nb < 4; nb++) {
        int c  = n0 + nb * 16 + ml;
        float bv = bias ? bias[c] : 0.0f;
        #pragma unroll
        for (int v = 0; v < 8; v++) {
            int r = m0 + wave * 16 + hi * 8 + v;
            float val = acc[nb][v] + bv;
            if (epilogue == 0) {
                C[(size_t)r * ldc + c] = val;
            } else {
                int b = r >> 11, l = r & (SEQ - 1);
                size_t o = ((size_t)(l * BATCH + b)) * HID + c;
                C[o] = resid[o] + val;
            }
        }
    }
}

// ---------------------------------------------------------------------------
// Prep / elementwise kernels
// ---------------------------------------------------------------------------
__global__ void xpose_cvt_kernel(const float* __restrict__ x,
                                 unsigned short* __restrict__ xbf) {
    size_t i = (size_t)blockIdx.x * 256 + threadIdx.x;
    if (i >= (size_t)NTOK * HID) return;
    int k = (int)(i & (HID - 1));
    int t = (int)(i >> 12);                // token in (b, l) order
    int b = t >> 11, l = t & (SEQ - 1);
    xbf[i] = f2bf(x[((size_t)l * BATCH + b) * HID + k]);
}

__global__ void wcvt_kernel(const float* __restrict__ src,
                            unsigned short* __restrict__ dst,
                            int N, int K, int Npad) {
    int i = blockIdx.x * 256 + threadIdx.x;
    if (i >= Npad * K) return;
    int n = i / K;
    dst[i] = (n < N) ? f2bf(src[i]) : (unsigned short)0;
}

__global__ void gelu_cvt_kernel(float* __restrict__ xm,
                                unsigned short* __restrict__ ub, int n) {
    int i = blockIdx.x * 256 + threadIdx.x;
    if (i >= n) return;
    float g = gelu_exact(xm[i]);
    xm[i] = g;
    ub[i] = f2bf(g);
}

__global__ void cvt_kernel(const float* __restrict__ s,
                           unsigned short* __restrict__ d, int n) {
    int i = blockIdx.x * 256 + threadIdx.x;
    if (i >= n) return;
    d[i] = f2bf(s[i]);
}

__global__ void mix_cvt_kernel(const float* __restrict__ xmid,
                               const float* __restrict__ hout,
                               const float* __restrict__ gate,
                               unsigned short* __restrict__ ub, int n) {
    int i = blockIdx.x * 256 + threadIdx.x;
    if (i >= n) return;
    float alpha = 1.0f / (1.0f + expf(-gate[0]));
    float v = (1.0f + alpha) * xmid[i] + (1.0f - alpha) * hout[i];
    ub[i] = f2bf(v);
}

// ---------------------------------------------------------------------------
// Depthwise causal conv (width 7) + SiLU over the xBC slice of zx.
// ---------------------------------------------------------------------------
__global__ void conv_silu_kernel(const float* __restrict__ zx,
                                 const float* __restrict__ conv_w,
                                 const float* __restrict__ conv_b,
                                 float* __restrict__ out) {
    int i = blockIdx.x * 256 + threadIdx.x;
    if (i >= NTOK * CONV_DIM) return;
    int c = i % CONV_DIM;
    int t = i / CONV_DIM;
    int b = t >> 11, l = t & (SEQ - 1);
    float acc = conv_b[c];
    #pragma unroll
    for (int j = 0; j < DCONV; j++) {
        int ll = l - (DCONV - 1) + j;
        if (ll >= 0)
            acc += conv_w[c * DCONV + j] *
                   zx[((size_t)(b * SEQ + ll)) * DPROJ_P + DIN + c];
    }
    out[i] = silu(acc);
}

// ---------------------------------------------------------------------------
// Selective-state scan. One block per (head, batch); dir 0 = fw, 1 = bw.
// 64 threads; thread p owns state row p (16 states in registers).
// ---------------------------------------------------------------------------
__global__ __launch_bounds__(64)
void ssm_scan_kernel(const float* __restrict__ zx,
                     const float* __restrict__ xBCc,
                     const float* __restrict__ dt_bias,
                     const float* __restrict__ A_log,
                     float* __restrict__ y_out, int dir) {
    const int nh = blockIdx.x;
    const int b  = blockIdx.y;
    const int p  = threadIdx.x;

    const float A    = -expf(A_log[nh]);
    const float bias = dt_bias[nh];

    float h[DST];
    #pragma unroll
    for (int n = 0; n < DST; n++) h[n] = 0.0f;

    __shared__ float BC[2 * DST];

    for (int s = 0; s < SEQ; s++) {
        int l = dir ? (SEQ - 1 - s) : s;
        size_t t = (size_t)b * SEQ + l;
        if (p < 2 * DST) BC[p] = xBCc[t * CONV_DIM + DIN + p];
        __syncthreads();

        float dtr = zx[t * DPROJ_P + DIN + CONV_DIM + dir * NH + nh] + bias;
        float dt  = softplus(dtr);
        float dA  = expf(dt * A);
        float xv  = xBCc[t * CONV_DIM + nh * HD + p];
        float dBx = dt * xv;
        float y = 0.0f;
        #pragma unroll
        for (int n = 0; n < DST; n++) {
            h[n] = dA * h[n] + dBx * BC[n];
            y   += h[n] * BC[DST + n];
        }
        y_out[t * DIN + nh * HD + p] = y;
        __syncthreads();
    }
}

// ---------------------------------------------------------------------------
// Shifted fw/bw combine + xs * (fcD + D) + gated RMSNorm -> bf16 for out-proj.
// One block per token; LDS float atomics for the fcD GEMV and RMS reduction.
// ---------------------------------------------------------------------------
__global__ __launch_bounds__(256)
void combine_norm_kernel(const float* __restrict__ zx,
                         const float* __restrict__ xBCc,
                         const float* __restrict__ y_fw,
                         const float* __restrict__ y_bw,
                         const float* __restrict__ fcD_w,
                         const float* __restrict__ D,
                         const float* __restrict__ norm_w,
                         unsigned short* __restrict__ ynorm_bf) {
    const int t = blockIdx.x;
    const int c = threadIdx.x;
    const int l = t & (SEQ - 1);

    __shared__ float red[NH + 1];
    if (c < NH + 1) red[c] = 0.0f;
    __syncthreads();

    float xs = xBCc[(size_t)t * CONV_DIM + c];
    #pragma unroll
    for (int hh = 0; hh < NH; hh++)
        atomicAdd(&red[hh], xs * fcD_w[hh * DIN + c]);
    __syncthreads();

    float yv = 0.0f;
    if (l > 0)       yv += y_fw[((size_t)t - 1) * DIN + c];
    if (l < SEQ - 1) yv += y_bw[((size_t)t + 1) * DIN + c];
    int hh = c >> 6;
    yv += xs * (red[hh] + D[hh]);

    float z  = zx[(size_t)t * DPROJ_P + c];
    float yz = yv * silu(z);

    atomicAdd(&red[NH], yz * yz);
    __syncthreads();
    float r = rsqrtf(red[NH] * (1.0f / DIN) + 1e-5f);
    ynorm_bf[(size_t)t * DIN + c] = f2bf(yz * r * norm_w[c]);
}

// ---------------------------------------------------------------------------
// Host-side orchestration
// ---------------------------------------------------------------------------
extern "C" void kernel_launch(void* const* d_in, const int* in_sizes, int n_in,
                              void* d_out, int out_size, void* d_ws, size_t ws_size,
                              hipStream_t stream) {
    (void)in_sizes; (void)n_in; (void)out_size; (void)ws_size;

    const float* x         = (const float*)d_in[0];
    const float* w_down    = (const float*)d_in[1];
    const float* b_down    = (const float*)d_in[2];
    const float* w_up      = (const float*)d_in[3];
    const float* b_up      = (const float*)d_in[4];
    const float* gate1     = (const float*)d_in[5];
    const float* hy_in_w   = (const float*)d_in[6];
    const float* hy_conv_w = (const float*)d_in[7];
    const float* hy_conv_b = (const float*)d_in[8];
    const float* hy_dt_b   = (const float*)d_in[9];
    const float* hy_A_log  = (const float*)d_in[10];
    const float* hy_D      = (const float*)d_in[11];
    const float* hy_fcD_w  = (const float*)d_in[12];
    const float* hy_norm_w = (const float*)d_in[13];
    const float* hy_out_w  = (const float*)d_in[14];
    float* out = (float*)d_out;

    // workspace carve-up
    char* w = (char*)d_ws;
    auto alloc = [&](size_t bytes) -> char* {
        char* p = w;
        w += (bytes + 255) & ~(size_t)255;
        return p;
    };
    unsigned short* xbf      = (unsigned short*)alloc((size_t)NTOK * HID * 2);
    unsigned short* wdown_bf = (unsigned short*)alloc((size_t)DM * HID * 2);
    unsigned short* wup_bf   = (unsigned short*)alloc((size_t)HID * DM * 2);
    unsigned short* winp_bf  = (unsigned short*)alloc((size_t)DPROJ_P * DM * 2);
    unsigned short* woutp_bf = (unsigned short*)alloc((size_t)DM * DIN * 2);
    float*          xmid     = (float*)alloc((size_t)NTOK * DM * 4);
    unsigned short* u_bf     = (unsigned short*)alloc((size_t)NTOK * DM * 2);
    float*          zx       = (float*)alloc((size_t)NTOK * DPROJ_P * 4);
    float*          xBCc     = (float*)alloc((size_t)NTOK * CONV_DIM * 4);
    float*          y_fw     = (float*)alloc((size_t)NTOK * DIN * 4);
    float*          y_bw     = (float*)alloc((size_t)NTOK * DIN * 4);
    unsigned short* ynorm    = (unsigned short*)alloc((size_t)NTOK * DIN * 2);
    float*          hout     = (float*)alloc((size_t)NTOK * DM * 4);

    const int nXE = NTOK * HID;         // 33.5M
    const int nMid = NTOK * DM;

    // 1) prep: bf16 conversions
    xpose_cvt_kernel<<<(nXE + 255) / 256, 256, 0, stream>>>(x, xbf);
    wcvt_kernel<<<(DM * HID + 255) / 256, 256, 0, stream>>>(w_down, wdown_bf, DM, HID, DM);
    wcvt_kernel<<<(HID * DM + 255) / 256, 256, 0, stream>>>(w_up, wup_bf, HID, DM, HID);
    wcvt_kernel<<<(DPROJ_P * DM + 255) / 256, 256, 0, stream>>>(hy_in_w, winp_bf, DPROJ, DM, DPROJ_P);
    wcvt_kernel<<<(DM * DIN + 255) / 256, 256, 0, stream>>>(hy_out_w, woutp_bf, DM, DIN, DM);

    // 2) down-proj GEMM + bias, then GELU (kept f32) + bf16 copy
    gemm_bf16_kernel<<<dim3(NTOK / 128, DM / 64), 256, 0, stream>>>(
        xbf, wdown_bf, b_down, xmid, NTOK, HID, DM, DM, 0, nullptr);
    gelu_cvt_kernel<<<(nMid + 255) / 256, 256, 0, stream>>>(xmid, u_bf, nMid);

    // 3) two hydra blocks
    for (int hblk = 0; hblk < 2; hblk++) {
        // in-proj: [8192,128] x [576,128]^T -> zx [8192,576]
        gemm_bf16_kernel<<<dim3(NTOK / 128, DPROJ_P / 64), 256, 0, stream>>>(
            u_bf, winp_bf, nullptr, zx, NTOK, DM, DPROJ_P, DPROJ_P, 0, nullptr);
        // causal conv + silu
        conv_silu_kernel<<<(NTOK * CONV_DIM + 255) / 256, 256, 0, stream>>>(
            zx, hy_conv_w, hy_conv_b, xBCc);
        // fw / bw scans
        ssm_scan_kernel<<<dim3(NH, BATCH), 64, 0, stream>>>(
            zx, xBCc, hy_dt_b, hy_A_log, y_fw, 0);
        ssm_scan_kernel<<<dim3(NH, BATCH), 64, 0, stream>>>(
            zx, xBCc, hy_dt_b, hy_A_log, y_bw, 1);
        // combine + fcD + gated RMSNorm -> bf16
        combine_norm_kernel<<<NTOK, 256, 0, stream>>>(
            zx, xBCc, y_fw, y_bw, hy_fcD_w, hy_D, hy_norm_w, ynorm);
        // out-proj: [8192,256] x [128,256]^T -> hout [8192,128]
        gemm_bf16_kernel<<<dim3(NTOK / 128, DM / 64), 256, 0, stream>>>(
            ynorm, woutp_bf, nullptr, hout, NTOK, DIN, DM, DM, 0, nullptr);
        if (hblk == 0)
            cvt_kernel<<<(nMid + 255) / 256, 256, 0, stream>>>(hout, u_bf, nMid);
    }

    // 4) gated mix:  h = (1+alpha)*xmid + (1-alpha)*hout  -> bf16
    mix_cvt_kernel<<<(nMid + 255) / 256, 256, 0, stream>>>(xmid, hout, gate1, u_bf, nMid);

    // 5) up-proj GEMM + b_up + residual with (l,b) re-transpose into d_out
    gemm_bf16_kernel<<<dim3(NTOK / 128, HID / 64), 256, 0, stream>>>(
        u_bf, wup_bf, b_up, out, NTOK, DM, HID, HID, 1, x);
}
